// droneDeliveryModel_9723805958639
// MI455X (gfx1250) — matrix operand order repaired and checked by reading, not compiled, not run on hardware
//
#include <hip/hip_runtime.h>
#include <hip/hip_fp16.h>

typedef __attribute__((ext_vector_type(16))) _Float16 v16h;
typedef __attribute__((ext_vector_type(8)))  float    v8f;

#define NEG_SLOPE 0.2f

// ---- order-preserving float<->uint mapping for atomic segment-max ----------
__device__ __forceinline__ unsigned mapOrd(float f) {
  unsigned b = __float_as_uint(f);
  return (b & 0x80000000u) ? ~b : (b | 0x80000000u);
}
__device__ __forceinline__ float unmapOrd(unsigned u) {
  unsigned b = (u & 0x80000000u) ? (u & 0x7fffffffu) : ~u;
  return __uint_as_float(b);
}

// ---- f32 [M,K] -> f16 [M,Kp] (zero pad K..Kp) ------------------------------
__global__ void cvt_a_f16(const float* __restrict__ src, _Float16* __restrict__ dst,
                          int M, int K, int Kp) {
  int t = blockIdx.x * blockDim.x + threadIdx.x;
  if (t >= M * Kp) return;
  int m = t / Kp, k = t - m * Kp;
  dst[t] = (k < K) ? (_Float16)src[(size_t)m * K + k] : (_Float16)0.0f;
}

// ---- W f32 [K,1024] -> W^T f16 [1024,Kp] (zero pad) ------------------------
__global__ void cvt_w_f16t(const float* __restrict__ W, _Float16* __restrict__ Wt,
                           int K, int Kp) {
  int t = blockIdx.x * blockDim.x + threadIdx.x;
  if (t >= 1024 * Kp) return;
  int n = t / Kp, k = t - n * Kp;
  Wt[t] = (k < K) ? (_Float16)W[(size_t)k * 1024 + n] : (_Float16)0.0f;
}

// ---- dual GEMM: Cl = A x Wl^T + bl, Cr = A x Wr^T + br, via v_wmma ---------
// One wave computes one 16x16 tile of BOTH outputs, sharing the A fragments.
// A fragment (16-bit A 16x32, ISA 7.12.2): row = tm*16 + lane%16,
//   lanes 0-15 hold K 0..15 contiguous, lanes 16-31 hold K 16..31 contiguous.
// B fragment (16-bit B 32x16, same k order as A): col = tn*16 + lane%16,
//   lanes 0-15 hold K 0..15 contiguous, lanes 16-31 hold K 16..31 contiguous.
__global__ void gemm2_wmma_bias(const _Float16* __restrict__ A,
                                const _Float16* __restrict__ Btl,
                                const _Float16* __restrict__ Btr,
                                const float* __restrict__ biasl,
                                const float* __restrict__ biasr,
                                float* __restrict__ Cl,
                                float* __restrict__ Cr,
                                int M, int Kp) {
  const int lane = threadIdx.x & 31;
  const int tile = blockIdx.x * (blockDim.x >> 5) + (threadIdx.x >> 5);
  const int ntn  = 64;                       // 1024 / 16
  const int tm   = tile / ntn, tn = tile % ntn;
  if (tm * 16 >= M) return;                  // wave-uniform guard (EXEC stays full)
  const int half = lane >> 4;                // 0 or 1
  const int l16  = lane & 15;
  const _Float16* ap  = A   + (size_t)(tm * 16 + l16) * Kp + half * 16;  // A pattern
  const _Float16* bpl = Btl + (size_t)(tn * 16 + l16) * Kp + half * 16;  // B pattern
  const _Float16* bpr = Btr + (size_t)(tn * 16 + l16) * Kp + half * 16;
  v8f accl = {}, accr = {};
  for (int kb = 0; kb < Kp; kb += 32) {
    v16h a, bl, br;
    ((float4*)&a)[0]  = *(const float4*)(ap + kb);        // K half*16 + 0..7
    ((float4*)&a)[1]  = *(const float4*)(ap + kb + 8);    // K half*16 + 8..15
    ((float4*)&bl)[0] = *(const float4*)(bpl + kb);       // K half*16 + 0..7
    ((float4*)&bl)[1] = *(const float4*)(bpl + kb + 8);   // K half*16 + 8..15
    ((float4*)&br)[0] = *(const float4*)(bpr + kb);
    ((float4*)&br)[1] = *(const float4*)(bpr + kb + 8);
    accl = __builtin_amdgcn_wmma_f32_16x16x32_f16(false, a, false, bl,
                                                  (short)0, accl, false, false);
    accr = __builtin_amdgcn_wmma_f32_16x16x32_f16(false, a, false, br,
                                                  (short)0, accr, false, false);
  }
  const int col   = tn * 16 + l16;
  const int rbase = tm * 16 + half * 8;
  const float bvl = biasl[col];
  const float bvr = biasr[col];
#pragma unroll
  for (int r = 0; r < 8; ++r) {
    Cl[(size_t)(rbase + r) * 1024 + col] = accl[r] + bvl;
    Cr[(size_t)(rbase + r) * 1024 + col] = accr[r] + bvr;
  }
}

// ---- attention logits: wave per (edge,head); fused leaky-relu/dot/seg-max --
__global__ void edge_alpha_k(const float* __restrict__ xl, const float* __restrict__ xr,
                             const int* __restrict__ ei, const float* __restrict__ att,
                             float* __restrict__ alpha, unsigned* __restrict__ maxb,
                             int E, int E2) {
  const int lane = threadIdx.x & 31;
  const int gw   = blockIdx.x * (blockDim.x >> 5) + (threadIdx.x >> 5);
  if (gw >= E2 * 8) return;
  const int e = gw >> 3, h = gw & 7;
  const int s = (e < E) ? ei[e]     : (e - E);
  const int d = (e < E) ? ei[E + e] : (e - E);
  float4 l  = ((const float4*)(xl  + (size_t)s * 1024 + h * 128))[lane];
  float4 r  = ((const float4*)(xr  + (size_t)d * 1024 + h * 128))[lane];
  float4 av = ((const float4*)(att + (size_t)h * 128))[lane];
  float v, sum = 0.0f;
  v = l.x + r.x; v = (v > 0.0f) ? v : NEG_SLOPE * v; sum += v * av.x;
  v = l.y + r.y; v = (v > 0.0f) ? v : NEG_SLOPE * v; sum += v * av.y;
  v = l.z + r.z; v = (v > 0.0f) ? v : NEG_SLOPE * v; sum += v * av.z;
  v = l.w + r.w; v = (v > 0.0f) ? v : NEG_SLOPE * v; sum += v * av.w;
#pragma unroll
  for (int off = 16; off > 0; off >>= 1) sum += __shfl_xor(sum, off, 32);
  if (lane == 0) {
    alpha[gw] = sum;
    atomicMax(&maxb[(size_t)d * 8 + h], mapOrd(sum));
  }
}

// ---- exp(alpha - max) and segment-sum denominators -------------------------
__global__ void edge_exp_k(float* __restrict__ alpha, const int* __restrict__ ei,
                           const unsigned* __restrict__ maxb, float* __restrict__ denom,
                           int E, int E2) {
  int t = blockIdx.x * blockDim.x + threadIdx.x;
  if (t >= E2 * 8) return;
  const int e = t >> 3, h = t & 7;
  const int d = (e < E) ? ei[E + e] : (e - E);
  float m = unmapOrd(maxb[(size_t)d * 8 + h]);
  float a = __expf(alpha[t] - m);
  alpha[t] = a;
  atomicAdd(&denom[(size_t)d * 8 + h], a);
}

// ---- weighted scatter aggregation: wave per (edge,head) --------------------
__global__ void edge_agg_k(const float* __restrict__ xl, const float* __restrict__ alpha,
                           const float* __restrict__ denom, const int* __restrict__ ei,
                           float* __restrict__ agg, int E, int E2) {
  const int lane = threadIdx.x & 31;
  const int gw   = blockIdx.x * (blockDim.x >> 5) + (threadIdx.x >> 5);
  if (gw >= E2 * 8) return;
  const int e = gw >> 3, h = gw & 7;
  const int s = (e < E) ? ei[e]     : (e - E);
  const int d = (e < E) ? ei[E + e] : (e - E);
  const float w = alpha[gw] / (denom[(size_t)d * 8 + h] + 1e-16f);
  float4 l = ((const float4*)(xl + (size_t)s * 1024 + h * 128))[lane];
  float* dp = agg + (size_t)d * 1024 + h * 128 + lane * 4;
  atomicAdd(dp + 0, w * l.x);
  atomicAdd(dp + 1, w * l.y);
  atomicAdd(dp + 2, w * l.z);
  atomicAdd(dp + 3, w * l.w);
}

// ---- head mean + bias + ReLU ----------------------------------------------
__global__ void head_mean_k(const float* __restrict__ agg, const float* __restrict__ bias,
                            float* __restrict__ hout, int N) {
  int t = blockIdx.x * blockDim.x + threadIdx.x;
  if (t >= N * 128) return;
  const int n = t >> 7, c = t & 127;
  float s = 0.0f;
#pragma unroll
  for (int h = 0; h < 8; ++h) s += agg[(size_t)n * 1024 + h * 128 + c];
  float v = s * 0.125f + bias[c];
  hout[t] = (v > 0.0f) ? v : 0.0f;
}

// ---- final projection on drone rows (rows 0..N_AGENTS-1) -------------------
__global__ void final_k(const float* __restrict__ h, const float* __restrict__ Wlin,
                        const float* __restrict__ blin, float* __restrict__ out,
                        int total) {
  int t = blockIdx.x * blockDim.x + threadIdx.x;
  if (t >= total) return;
  const int n = t >> 3, c = t & 7;
  float s = blin[c];
  const float* hp = h + (size_t)n * 128;
#pragma unroll 8
  for (int k = 0; k < 128; ++k) s += hp[k] * Wlin[k * 8 + c];
  out[t] = s;
}

extern "C" void kernel_launch(void* const* d_in, const int* in_sizes, int n_in,
                              void* d_out, int out_size, void* d_ws, size_t ws_size,
                              hipStream_t stream) {
  const float* x    = (const float*)d_in[0];
  const int*   ei   = (const int*)  d_in[1];
  const float* Wl1  = (const float*)d_in[2];
  const float* bl1  = (const float*)d_in[3];
  const float* Wr1  = (const float*)d_in[4];
  const float* br1  = (const float*)d_in[5];
  const float* att1 = (const float*)d_in[6];
  const float* b1   = (const float*)d_in[7];
  const float* Wl2  = (const float*)d_in[8];
  const float* bl2  = (const float*)d_in[9];
  const float* Wr2  = (const float*)d_in[10];
  const float* br2  = (const float*)d_in[11];
  const float* att2 = (const float*)d_in[12];
  const float* b2   = (const float*)d_in[13];
  const float* Wlin = (const float*)d_in[14];
  const float* blin = (const float*)d_in[15];

  const int N  = in_sizes[0] / 16;   // 10000
  const int E  = in_sizes[1] / 2;    // 160000
  const int E2 = E + N;              // with self-loops

  // ---- carve workspace (~96 MB) ----
  char* ws = (char*)d_ws;
  size_t off = 0;
  auto carve = [&](size_t bytes) -> void* {
    void* p = ws + off;
    off = (off + bytes + 255) & ~(size_t)255;
    return p;
  };
  float*    xl     = (float*)   carve((size_t)N * 1024 * 4);
  float*    xr     = (float*)   carve((size_t)N * 1024 * 4);  // reused as agg
  float*    alpha  = (float*)   carve((size_t)E2 * 8 * 4);
  unsigned* maxb   = (unsigned*)carve((size_t)N * 8 * 4);
  float*    denom  = (float*)   carve((size_t)N * 8 * 4);
  float*    hbuf   = (float*)   carve((size_t)N * 128 * 4);
  _Float16* a16    = (_Float16*)carve((size_t)N * 128 * 2);
  _Float16* wtl16  = (_Float16*)carve((size_t)1024 * 128 * 2);
  _Float16* wtr16  = (_Float16*)carve((size_t)1024 * 128 * 2);
  (void)ws_size; (void)n_in;

  const int TB = 256;
  auto blocks = [&](int n) { return (n + TB - 1) / TB; };

  auto run_layer = [&](const float* in, int K, int Kp,
                       const float* Wl, const float* bl,
                       const float* Wr, const float* br,
                       const float* att, const float* bias, float* hout) {
    cvt_a_f16 <<<blocks(N * Kp),    TB, 0, stream>>>(in, a16, N, K, Kp);
    cvt_w_f16t<<<blocks(1024 * Kp), TB, 0, stream>>>(Wl, wtl16, K, Kp);
    cvt_w_f16t<<<blocks(1024 * Kp), TB, 0, stream>>>(Wr, wtr16, K, Kp);
    const int tiles = (N / 16) * 64;
    gemm2_wmma_bias<<<(tiles + 7) / 8, TB, 0, stream>>>(a16, wtl16, wtr16,
                                                        bl, br, xl, xr, N, Kp);

    hipMemsetAsync(maxb,  0, (size_t)N * 8 * 4, stream);
    hipMemsetAsync(denom, 0, (size_t)N * 8 * 4, stream);
    edge_alpha_k<<<(E2 * 8 + 7) / 8, TB, 0, stream>>>(xl, xr, ei, att, alpha, maxb, E, E2);
    edge_exp_k  <<<blocks(E2 * 8),   TB, 0, stream>>>(alpha, ei, maxb, denom, E, E2);

    hipMemsetAsync(xr, 0, (size_t)N * 1024 * 4, stream);   // xr becomes agg
    edge_agg_k <<<(E2 * 8 + 7) / 8, TB, 0, stream>>>(xl, alpha, denom, ei, xr, E, E2);
    head_mean_k<<<blocks(N * 128),  TB, 0, stream>>>(xr, bias, hout, N);
  };

  run_layer(x,    16, 32,   Wl1, bl1, Wr1, br1, att1, b1, hbuf);
  run_layer(hbuf, 128, 128, Wl2, bl2, Wr2, br2, att2, b2, hbuf);

  final_k<<<blocks(out_size), TB, 0, stream>>>(hbuf, Wlin, blin, (float*)d_out, out_size);
}
